// HardNegativeTripletLoss_7438883357145
// MI455X (gfx1250) — matrix operand order repaired and checked by reading, not compiled, not run on hardware
//
#include <hip/hip_runtime.h>
#include <hip/hip_bf16.h>

// ---------------- problem constants (match reference) ----------------
#define BATCH     8192
#define DIM       512
#define INSTANCES 4
#define MARGIN    0.3f
#define K_HARD    4094          // (B - INSTANCES) * 0.5
#define EPS       1e-8f

#define NBINS     1024          // histogram bins over [-1, 1]
#define ROWS      16            // rows per strip (one workgroup)
#define THREADS   256           // 8 waves of 32
#define RDW       (DIM / 2)     // row length in uint (2 bf16 each) = 256

typedef __attribute__((ext_vector_type(16))) __bf16 v16bf;
typedef __attribute__((ext_vector_type(8)))  float  v8f;

union BF16x16 { v16bf v; uint4 q[2]; };

// float -> bf16 with round-to-nearest-even
static __device__ __forceinline__ unsigned int f2bf(float f) {
    unsigned int u = __float_as_uint(f);
    u += 0x7FFFu + ((u >> 16) & 1u);
    return u >> 16;
}

// order-preserving float<->uint key (uint ascending == float ascending)
static __device__ __forceinline__ unsigned int fkey(float f) {
    unsigned int u = __float_as_uint(f);
    return (u & 0x80000000u) ? ~u : (u | 0x80000000u);
}
static __device__ __forceinline__ float fdecode(unsigned int k) {
    unsigned int u = (k & 0x80000000u) ? (k & 0x7FFFFFFFu) : ~k;
    return __uint_as_float(u);
}

// ---------------- kernel 0: zero the scalar output ----------------
__global__ void zero_out_kernel(float* out) {
    if (threadIdx.x == 0 && blockIdx.x == 0) out[0] = 0.0f;
}

// ---------------- kernel 1: L2-normalize rows, emit bf16 panel ----------------
// one wave per row; each lane handles 16 floats (4x float4)
__global__ void normalize_kernel(const float* __restrict__ e,
                                 unsigned int* __restrict__ xn /* packed 2xbf16 */) {
    const int wave = threadIdx.x >> 5;
    const int lane = threadIdx.x & 31;
    const int row  = blockIdx.x * 8 + wave;

    const float4* src = (const float4*)(e + (size_t)row * DIM);
    float4 v[4];
    float ss = 0.0f;
#pragma unroll
    for (int i = 0; i < 4; ++i) {
        v[i] = src[lane + 32 * i];
        ss += v[i].x * v[i].x + v[i].y * v[i].y + v[i].z * v[i].z + v[i].w * v[i].w;
    }
#pragma unroll
    for (int m = 16; m >= 1; m >>= 1) ss += __shfl_xor(ss, m, 32);

    const float inv = 1.0f / fmaxf(sqrtf(ss), EPS);

    unsigned int* dst = xn + (size_t)row * RDW;
#pragma unroll
    for (int i = 0; i < 4; ++i) {
        unsigned int lo = f2bf(v[i].x * inv) | (f2bf(v[i].y * inv) << 16);
        unsigned int hi = f2bf(v[i].z * inv) | (f2bf(v[i].w * inv) << 16);
        dst[(lane + 32 * i) * 2 + 0] = lo;
        dst[(lane + 32 * i) * 2 + 1] = hi;
    }
}

// ---------------- classify one 16x16 accumulator tile ----------------
static __device__ __forceinline__ void classify_tile(
    const v8f& acc, int n0, int row0, int lane,
    const int* __restrict__ pids, const int* spids,
    unsigned int* posmin, unsigned int* cnt, float* fsum) {

    const int mbase = (lane >> 4) * 8;          // C layout: VGPR j -> M = j + 8*(lane/16)
    const int col   = n0 + (lane & 15);         // N = lane % 16
    const int cp    = pids[col];                // 32KB array, L2-hot
#pragma unroll
    for (int j = 0; j < 8; ++j) {
        const int m   = mbase + j;
        const float s = acc[j];
        if (spids[m] == cp) {
            if (row0 + m != col)                // exclude self
                atomicMin(&posmin[m], fkey(s));
        } else {
            int bin = (int)floorf((s + 1.0f) * (float)(NBINS / 2));
            bin = bin < 0 ? 0 : (bin > NBINS - 1 ? NBINS - 1 : bin);
            atomicAdd(&cnt[m * NBINS + bin], 1u);       // ds_add_u32
            atomicAdd(&fsum[m * NBINS + bin], s);       // ds_add_f32
        }
    }
}

// ---------------- kernel 2: fused sim-GEMM + hard mining + loss ----------------
// grid = BATCH/ROWS workgroups
// dynamic LDS layout: atile[ROWS*RDW] | cnt[ROWS*NBINS] | fsum[ROWS*NBINS] | posmin[ROWS] | spids[ROWS]
__global__ __launch_bounds__(THREADS)
__attribute__((amdgpu_waves_per_eu(1)))
void triplet_kernel(const unsigned int* __restrict__ xn,
                    const int* __restrict__ pids,
                    float* __restrict__ out) {
    extern __shared__ char smem[];
    unsigned int* atile  = (unsigned int*)smem;                   // 16 KB (A strip, bf16)
    unsigned int* cnt    = atile + ROWS * RDW;                    // [ROWS][NBINS]
    float*        fsum   = (float*)(cnt + ROWS * NBINS);          // [ROWS][NBINS]
    unsigned int* posmin = (unsigned int*)(fsum + ROWS * NBINS);  // [ROWS]
    int*          spids  = (int*)(posmin + ROWS);                 // [ROWS]

    const int row0 = blockIdx.x * ROWS;

    // stage A strip (16 contiguous rows of bf16 panel) into LDS: 16 KB
    {
        const uint4* srcA = (const uint4*)(xn + (size_t)row0 * RDW);
        uint4*       dstA = (uint4*)atile;
        for (int i = threadIdx.x; i < ROWS * RDW / 4; i += THREADS) dstA[i] = srcA[i];
    }
    for (int i = threadIdx.x; i < ROWS * NBINS; i += THREADS) { cnt[i] = 0u; fsum[i] = 0.0f; }
    if (threadIdx.x < ROWS) {
        posmin[threadIdx.x] = 0xFFFFFFFFu;
        spids[threadIdx.x]  = pids[row0 + threadIdx.x];
    }
    __syncthreads();

    const int wave = threadIdx.x >> 5;
    const int lane = threadIdx.x & 31;
    const int m    = lane & 15;
    // bf16 A layout: lane holds 16 contiguous K at k0 + 16*(lane/16), M = lane%16
    const unsigned int* aLds = atile + m * RDW + (lane >> 4) * 8;   // uint units, in LDS
    // bf16 B layout: lane holds 16 contiguous K at k0 + 16*(lane/16), N = lane%16
    const size_t bsel = (size_t)(lane >> 4) * 8;                    // uint units

    // each wave produces two 16-column tiles per iteration; B is double-buffered so
    // loads for K-step k+1 are in flight while the WMMAs of step k execute.
    for (int pair = wave; pair < BATCH / 32; pair += 8) {
        const int n0 = pair * 32;
        const unsigned int* brow0 = xn + (size_t)(n0 + m)      * RDW + bsel;
        const unsigned int* brow1 = xn + (size_t)(n0 + 16 + m) * RDW + bsel;

        BF16x16 b0c, b1c, b0n, b1n;
        b0c.q[0] = *(const uint4*)(brow0);
        b0c.q[1] = *(const uint4*)(brow0 + 4);
        b1c.q[0] = *(const uint4*)(brow1);
        b1c.q[1] = *(const uint4*)(brow1 + 4);

        v8f acc0 = {}; v8f acc1 = {};
#pragma unroll
        for (int k = 0; k < 16; ++k) {                // 32 bf16 of K per step
            const int k0 = k * 16;
            if (k < 15) {                             // prefetch next K chunk of B
                b0n.q[0] = *(const uint4*)(brow0 + k0 + 16);
                b0n.q[1] = *(const uint4*)(brow0 + k0 + 20);
                b1n.q[0] = *(const uint4*)(brow1 + k0 + 16);
                b1n.q[1] = *(const uint4*)(brow1 + k0 + 20);
            }
            BF16x16 a;                                 // A from LDS (ds_load_b128 x2)
            a.q[0] = *(const uint4*)(aLds + k0);       // K chunk +0..7 (or +16..23)
            a.q[1] = *(const uint4*)(aLds + k0 + 4);   // K chunk +8..15 (or +24..31)
            acc0 = __builtin_amdgcn_wmma_f32_16x16x32_bf16(
                       false, a.v, false, b0c.v, (short)0, acc0, false, false);
            acc1 = __builtin_amdgcn_wmma_f32_16x16x32_bf16(
                       false, a.v, false, b1c.v, (short)0, acc1, false, false);
            b0c = b0n; b1c = b1n;                      // rotate (renamed by unroll)
        }
        classify_tile(acc0, n0,      row0, lane, pids, spids, posmin, cnt, fsum);
        classify_tile(acc1, n0 + 16, row0, lane, pids, spids, posmin, cnt, fsum);
    }
    __syncthreads();

    // per-row selection: top-K_HARD mean via histogram scan (threads 0..15)
    if (threadIdx.x < ROWS) {
        const int r = threadIdx.x;
        unsigned int cumC = 0; float cumS = 0.0f; float topk = 0.0f;
        for (int b = NBINS - 1; b >= 0; --b) {
            const unsigned int c = cnt[r * NBINS + b];
            const float        s = fsum[r * NBINS + b];
            if (cumC + c >= (unsigned)K_HARD) {
                const float avg = s / (float)c;             // c > 0 here
                topk = cumS + (float)(K_HARD - cumC) * avg;
                break;
            }
            cumC += c; cumS += s;
        }
        const float sim_an = topk / (float)K_HARD;
        const unsigned int key = posmin[r];
        const float sim_ap = (key != 0xFFFFFFFFu) ? fdecode(key) : 1.0f;
        const float rl = fmaxf(sim_an - sim_ap + MARGIN, 0.0f);
        atomicAdd(out, rl * (1.0f / (float)BATCH));         // global_atomic_add_f32
    }
}

// ---------------- launcher ----------------
extern "C" void kernel_launch(void* const* d_in, const int* in_sizes, int n_in,
                              void* d_out, int out_size, void* d_ws, size_t ws_size,
                              hipStream_t stream) {
    const float* embeds = (const float*)d_in[0];
    const int*   pids   = (const int*)d_in[1];
    float*       out    = (float*)d_out;
    unsigned int* xn    = (unsigned int*)d_ws;   // BATCH*DIM bf16 = 8.4 MB

    (void)in_sizes; (void)n_in; (void)out_size; (void)ws_size;

    const size_t lds_bytes = (size_t)ROWS * RDW * 4      // A strip
                           + (size_t)ROWS * NBINS * 8    // histograms
                           + ROWS * 8;                   // posmin + spids
    (void)hipFuncSetAttribute((const void*)triplet_kernel,
                              hipFuncAttributeMaxDynamicSharedMemorySize,
                              (int)lds_bytes);

    zero_out_kernel<<<1, 64, 0, stream>>>(out);
    normalize_kernel<<<BATCH / 8, THREADS, 0, stream>>>(embeds, xn);
    triplet_kernel<<<BATCH / ROWS, THREADS, lds_bytes, stream>>>(xn, pids, out);
}
